// NormBinModel_43817256354081
// MI455X (gfx1250) — 8-bit matrix operand order repaired and checked by reading, compiled, not run on hardware
//
#include <hip/hip_runtime.h>
#include <hip/hip_bf16.h>

typedef __attribute__((ext_vector_type(8))) int v8i;
typedef __attribute__((ext_vector_type(4))) int v4i;
typedef __attribute__((ext_vector_type(2))) int v2i;

#define EPSF 1e-5f

__device__ __forceinline__ signed char bsign(float t) {
  return t >= 0.0f ? (signed char)1 : (signed char)(-1);
}

__global__ void pack_conv_w(const float* __restrict__ w, signed char* __restrict__ wb,
                            int O, int C) {
  int idx = blockIdx.x * blockDim.x + threadIdx.x;
  int total = O * C * 9;
  if (idx >= total) return;
  int dx = idx % 3;
  int t  = idx / 3;
  int dy = t % 3;  t /= 3;
  int c  = t % C;
  int o  = t / C;
  float f = w[idx];
  wb[(size_t)o * (9 * C) + (dy * 3 + dx) * C + c] = (f >= 0.0f) ? (signed char)1 : (signed char)-1;
}

__global__ void pack_lin_w(const float* __restrict__ w, signed char* __restrict__ wb,
                           int N, int Npad, int K, int perm) {
  int idx = blockIdx.x * blockDim.x + threadIdx.x;
  int total = Npad * K;
  if (idx >= total) return;
  int k = idx % K;
  int o = idx / K;
  signed char v = 0;
  if (o < N) {
    float f = w[(size_t)o * K + k];
    v = (f >= 0.0f) ? (signed char)1 : (signed char)-1;
  }
  int kp = k;
  if (perm) {
    int c = k / 16;
    int pos = k % 16;
    kp = pos * 512 + c;
  }
  wb[(size_t)o * K + kp] = v;
}

__global__ void conv1_kernel(const float* __restrict__ x, const float* __restrict__ w,
                             const float* __restrict__ bias,
                             const float* __restrict__ g, const float* __restrict__ bb,
                             const float* __restrict__ m, const float* __restrict__ v,
                             signed char* __restrict__ out,
                             int B, int H, int W, int O) {
  int idx = blockIdx.x * blockDim.x + threadIdx.x;
  int total = B * H * W * O;
  if (idx >= total) return;
  int o = idx % O;
  int p = idx / O;
  int xw = p % W;
  int t = p / W;
  int yh = t % H;
  int b = t / H;
  float acc = bias[o];
  for (int c = 0; c < 3; ++c) {
    const float* xp = x + ((size_t)b * 3 + c) * H * W;
    const float* wp = w + ((size_t)o * 3 + c) * 9;
#pragma unroll
    for (int dy = 0; dy < 3; ++dy) {
      int iy = yh + dy - 1;
      if (iy < 0 || iy >= H) continue;
#pragma unroll
      for (int dx = 0; dx < 3; ++dx) {
        int ix = xw + dx - 1;
        if (ix < 0 || ix >= W) continue;
        float wv = (wp[dy * 3 + dx] >= 0.0f) ? 1.0f : -1.0f;
        acc += xp[iy * W + ix] * wv;
      }
    }
  }
  float s = g[o] * rsqrtf(v[o] + EPSF);
  out[idx] = bsign((acc - m[o]) * s + bb[o]);
}

template <int C>
__global__ void __launch_bounds__(128)
binconv_wmma(const signed char* __restrict__ in, const signed char* __restrict__ wgt,
             const float* __restrict__ bias,
             const float* __restrict__ bg, const float* __restrict__ bnb,
             const float* __restrict__ bnm, const float* __restrict__ bnv,
             short* __restrict__ out_i16, signed char* __restrict__ out_s,
             int B, int H, int W, int O, int sign_mode) {
  constexpr int MT = 2, NT = 4;
  constexpr int K9 = 9 * C;
  constexpr int CPT = C / 64;
  constexpr int NC = 9 * CPT;
  const int lane = threadIdx.x & 31;
  const int wave = threadIdx.x >> 5;
  const int ngroups = O >> 6;
  const int mgroups = (B * H * W) >> 5;
  const int tile = blockIdx.x * 4 + wave;
  if (tile >= mgroups * ngroups) return;
  const int mg = tile / ngroups;
  const int ng = tile % ngroups;
  const int mrow = lane & 15;
  const int khalf = lane >> 4;

  int xw[MT], yh[MT], bpos[MT];
#pragma unroll
  for (int mt = 0; mt < MT; ++mt) {
    int p = (mg * MT + mt) * 16 + mrow;
    xw[mt] = p % W;
    int t0 = p / W;
    yh[mt] = t0 % H;
    bpos[mt] = t0 / H;
  }
  int oc[NT];
  const signed char* wrow[NT];
#pragma unroll
  for (int nt = 0; nt < NT; ++nt) {
    oc[nt] = (ng * NT + nt) * 16 + mrow;
    wrow[nt] = wgt + (size_t)oc[nt] * K9 + khalf * 16;
  }

  v8i acc[MT][NT] = {};

  auto load_chunk = [&](int i, v8i (&a)[MT], v8i (&bf)[NT]) {
    const int tap = i / CPT;
    const int c0 = (i % CPT) * 64;
    const int dy = tap / 3, dx = tap % 3;
#pragma unroll
    for (int mt = 0; mt < MT; ++mt) {
      int iy = yh[mt] + dy - 1;
      int ix = xw[mt] + dx - 1;
      bool ok = (iy >= 0) && (iy < H) && (ix >= 0) && (ix < W);
      const signed char* q =
          in + (((long)bpos[mt] * H + iy) * W + ix) * (long)C + khalf * 16 + c0;
      a[mt] = (v8i){};
      if (ok) {
        v2i u0 = *(const v2i*)(q);
        v2i u1 = *(const v2i*)(q + 8);
        v2i u2 = *(const v2i*)(q + 32);
        v2i u3 = *(const v2i*)(q + 40);
        a[mt][0] = u0.x; a[mt][1] = u0.y; a[mt][2] = u1.x; a[mt][3] = u1.y;
        a[mt][4] = u2.x; a[mt][5] = u2.y; a[mt][6] = u3.x; a[mt][7] = u3.y;
      }
    }
    const int kb = i * 64;
#pragma unroll
    for (int nt = 0; nt < NT; ++nt) {
      const signed char* q = wrow[nt] + kb;
      v4i w0 = *(const v4i*)(q);
      v4i w1 = *(const v4i*)(q + 32);
      bf[nt][0] = w0.x; bf[nt][1] = w0.y; bf[nt][2] = w0.z; bf[nt][3] = w0.w;
      bf[nt][4] = w1.x; bf[nt][5] = w1.y; bf[nt][6] = w1.z; bf[nt][7] = w1.w;
    }
  };
  auto domul = [&](v8i (&a)[MT], v8i (&bf)[NT]) {
#pragma unroll
    for (int mt = 0; mt < MT; ++mt)
#pragma unroll
      for (int nt = 0; nt < NT; ++nt)
        acc[mt][nt] = __builtin_amdgcn_wmma_i32_16x16x64_iu8(
            true, a[mt], true, bf[nt], acc[mt][nt], false, false);
  };

  v8i a0[MT], b0[NT], a1[MT], b1[NT];
  load_chunk(0, a0, b0);
#pragma unroll
  for (int i = 0; i < NC; i += 2) {
    load_chunk(i + 1, a1, b1);
    domul(a0, b0);
    if (i + 2 < NC) load_chunk(i + 2, a0, b0);
    domul(a1, b1);
  }

  const int hi = khalf * 8;
#pragma unroll
  for (int nt = 0; nt < NT; ++nt) {
    float bia = 0.0f, sc = 0.0f, mm = 0.0f, bo = 0.0f;
    if (sign_mode) {
      bia = bias[oc[nt]];
      sc = bg[oc[nt]] * rsqrtf(bnv[oc[nt]] + EPSF);
      mm = bnm[oc[nt]];
      bo = bnb[oc[nt]];
    }
#pragma unroll
    for (int mt = 0; mt < MT; ++mt) {
#pragma unroll
      for (int r = 0; r < 8; ++r) {
        int pm = (mg * MT + mt) * 16 + hi + r;
        int ox = pm % W;
        int pt = pm / W;
        int oy = pt % H;
        int ob = pt / H;
        size_t oidx = ((size_t)((ob * H + oy) * W + ox)) * O + oc[nt];
        if (sign_mode) {
          float vv = (float)acc[mt][nt][r] + bia;
          out_s[oidx] = bsign((vv - mm) * sc + bo);
        } else {
          out_i16[oidx] = (short)acc[mt][nt][r];
        }
      }
    }
  }
}

__global__ void __launch_bounds__(128)
bingemm_wmma(const signed char* __restrict__ a, const signed char* __restrict__ wgt,
             const float* __restrict__ bias,
             const float* __restrict__ bg, const float* __restrict__ bnb,
             const float* __restrict__ bnm, const float* __restrict__ bnv,
             float* __restrict__ out_f, signed char* __restrict__ out_s,
             int M, int K, int Npad, int Nvalid, int sign_mode) {
  constexpr int MT = 2, NT = 4;
  const int lane = threadIdx.x & 31;
  const int wave = threadIdx.x >> 5;
  const int ngroups = Npad >> 6;
  const int mgroups = M >> 5;
  const int tile = blockIdx.x * 4 + wave;
  if (tile >= mgroups * ngroups) return;
  const int mg = tile / ngroups;
  const int ng = tile % ngroups;
  const int mrow = lane & 15;
  const int khalf = lane >> 4;

  const signed char* ap[MT];
#pragma unroll
  for (int mt = 0; mt < MT; ++mt) {
    int row = (mg * MT + mt) * 16 + mrow;
    ap[mt] = a + (size_t)row * K + khalf * 16;
  }
  int oc[NT];
  const signed char* wp[NT];
#pragma unroll
  for (int nt = 0; nt < NT; ++nt) {
    oc[nt] = (ng * NT + nt) * 16 + mrow;
    wp[nt] = wgt + (size_t)oc[nt] * K + khalf * 16;
  }

  v8i acc[MT][NT] = {};

  auto load_chunk = [&](int i, v8i (&av)[MT], v8i (&bf)[NT]) {
    const int k0 = i * 64;
#pragma unroll
    for (int mt = 0; mt < MT; ++mt) {
      const signed char* q = ap[mt] + k0;
      v2i u0 = *(const v2i*)(q);
      v2i u1 = *(const v2i*)(q + 8);
      v2i u2 = *(const v2i*)(q + 32);
      v2i u3 = *(const v2i*)(q + 40);
      av[mt][0] = u0.x; av[mt][1] = u0.y; av[mt][2] = u1.x; av[mt][3] = u1.y;
      av[mt][4] = u2.x; av[mt][5] = u2.y; av[mt][6] = u3.x; av[mt][7] = u3.y;
    }
#pragma unroll
    for (int nt = 0; nt < NT; ++nt) {
      const signed char* q = wp[nt] + k0;
      v4i w0 = *(const v4i*)(q);
      v4i w1 = *(const v4i*)(q + 32);
      bf[nt][0] = w0.x; bf[nt][1] = w0.y; bf[nt][2] = w0.z; bf[nt][3] = w0.w;
      bf[nt][4] = w1.x; bf[nt][5] = w1.y; bf[nt][6] = w1.z; bf[nt][7] = w1.w;
    }
  };
  auto domul = [&](v8i (&av)[MT], v8i (&bf)[NT]) {
#pragma unroll
    for (int mt = 0; mt < MT; ++mt)
#pragma unroll
      for (int nt = 0; nt < NT; ++nt)
        acc[mt][nt] = __builtin_amdgcn_wmma_i32_16x16x64_iu8(
            true, av[mt], true, bf[nt], acc[mt][nt], false, false);
  };

  const int NCc = K >> 6;
  v8i a0[MT], b0[NT], a1[MT], b1[NT];
  load_chunk(0, a0, b0);
  int i = 0;
  for (; i + 2 < NCc; i += 2) {
    load_chunk(i + 1, a1, b1);
    domul(a0, b0);
    load_chunk(i + 2, a0, b0);
    domul(a1, b1);
  }
  load_chunk(i + 1, a1, b1);
  domul(a0, b0);
  domul(a1, b1);

  const int hi = khalf * 8;
#pragma unroll
  for (int nt = 0; nt < NT; ++nt) {
    float bia = 0.0f, sc = 0.0f, mm = 0.0f, bo = 0.0f;
    if (oc[nt] < Nvalid) {
      bia = bias[oc[nt]];
      sc = bg[oc[nt]] * rsqrtf(bnv[oc[nt]] + EPSF);
      mm = bnm[oc[nt]];
      bo = bnb[oc[nt]];
    }
#pragma unroll
    for (int mt = 0; mt < MT; ++mt) {
#pragma unroll
      for (int r = 0; r < 8; ++r) {
        int orow = (mg * MT + mt) * 16 + hi + r;
        float vv = (float)acc[mt][nt][r] + bia;
        float tb = (vv - mm) * sc + bo;
        if (sign_mode) {
          out_s[(size_t)orow * Npad + oc[nt]] = bsign(tb);
        } else {
          out_f[(size_t)orow * Npad + oc[nt]] = tb;
        }
      }
    }
  }
}

__global__ void pool_bn_sign(const short* __restrict__ in, signed char* __restrict__ out,
                             const float* __restrict__ bias,
                             const float* __restrict__ g, const float* __restrict__ bb,
                             const float* __restrict__ m, const float* __restrict__ v,
                             int B, int H, int W, int O) {
  int Ho = H >> 1, Wo = W >> 1;
  int idx = blockIdx.x * blockDim.x + threadIdx.x;
  int total = B * Ho * Wo * O;
  if (idx >= total) return;
  int o = idx % O;
  int p = idx / O;
  int x = p % Wo;  p /= Wo;
  int y = p % Ho;
  int b = p / Ho;
  size_t base = ((size_t)((b * H + 2 * y) * W + 2 * x)) * O + o;
  int v0 = in[base];
  int v1 = in[base + O];
  int v2 = in[base + (size_t)W * O];
  int v3 = in[base + (size_t)W * O + O];
  int mx = max(max(v0, v1), max(v2, v3));
  float vv = (float)mx + bias[o];
  float s = g[o] * rsqrtf(v[o] + EPSF);
  out[idx] = bsign((vv - m[o]) * s + bb[o]);
}

__global__ void logsoftmax_kernel(const float* __restrict__ in, float* __restrict__ out,
                                  int M, int N, int stride) {
  int r = blockIdx.x * blockDim.x + threadIdx.x;
  if (r >= M) return;
  const float* p = in + (size_t)r * stride;
  float mx = p[0];
  for (int i = 1; i < N; ++i) mx = fmaxf(mx, p[i]);
  float s = 0.0f;
  for (int i = 0; i < N; ++i) s += __expf(p[i] - mx);
  float l = __logf(s);
  for (int i = 0; i < N; ++i) out[(size_t)r * N + i] = p[i] - mx - l;
}

static inline int divup(int a, int b) { return (a + b - 1) / b; }

extern "C" void kernel_launch(void* const* d_in, const int* in_sizes, int n_in,
                              void* d_out, int out_size, void* d_ws, size_t ws_size,
                              hipStream_t stream) {
  (void)in_sizes; (void)n_in; (void)out_size; (void)ws_size;
  const int B = 512;

  const float* x = (const float*)d_in[0];
  const float* cw[7];  const float* cb[7];
  for (int i = 1; i <= 6; ++i) { cw[i] = (const float*)d_in[2 * i - 1]; cb[i] = (const float*)d_in[2 * i]; }
  auto bn = [&](int i, int j) { return (const float*)d_in[13 + (i - 1) * 4 + j]; };
  const float* lw1 = (const float*)d_in[49]; const float* lb1 = (const float*)d_in[50];
  const float* lw2 = (const float*)d_in[51]; const float* lb2 = (const float*)d_in[52];
  const float* lw3 = (const float*)d_in[53]; const float* lb3 = (const float*)d_in[54];

  char* ws = (char*)d_ws;
  const size_t SZ_F = (size_t)B * 32 * 32 * 128 * 2;
  const size_t SZ_A = (size_t)B * 32 * 32 * 128;
  const size_t SZ_B = (size_t)B * 16 * 16 * 128;
  short*       Fbuf = (short*)(ws);
  signed char* Abuf = (signed char*)(ws + SZ_F);
  signed char* Bbuf = (signed char*)(ws + SZ_F + SZ_A);
  char* wr = ws + SZ_F + SZ_A + SZ_B;
  signed char* wb2 = (signed char*)wr;                 wr += (size_t)128 * 1152;
  signed char* wb3 = (signed char*)wr;                 wr += (size_t)256 * 1152;
  signed char* wb4 = (signed char*)wr;                 wr += (size_t)256 * 2304;
  signed char* wb5 = (signed char*)wr;                 wr += (size_t)512 * 2304;
  signed char* wb6 = (signed char*)wr;                 wr += (size_t)512 * 4608;
  signed char* wl1p = (signed char*)wr;                wr += (size_t)1024 * 8192;
  signed char* wl2p = (signed char*)wr;                wr += (size_t)1024 * 1024;
  signed char* wl3p = (signed char*)wr;                wr += (size_t)64 * 1024;
  float*       logits = (float*)wr;

  pack_conv_w<<<divup(128 * 128 * 9, 256), 256, 0, stream>>>(cw[2], wb2, 128, 128);
  pack_conv_w<<<divup(256 * 128 * 9, 256), 256, 0, stream>>>(cw[3], wb3, 256, 128);
  pack_conv_w<<<divup(256 * 256 * 9, 256), 256, 0, stream>>>(cw[4], wb4, 256, 256);
  pack_conv_w<<<divup(512 * 256 * 9, 256), 256, 0, stream>>>(cw[5], wb5, 512, 256);
  pack_conv_w<<<divup(512 * 512 * 9, 256), 256, 0, stream>>>(cw[6], wb6, 512, 512);
  pack_lin_w<<<divup(1024 * 8192, 256), 256, 0, stream>>>(lw1, wl1p, 1024, 1024, 8192, 1);
  pack_lin_w<<<divup(1024 * 1024, 256), 256, 0, stream>>>(lw2, wl2p, 1024, 1024, 1024, 0);
  pack_lin_w<<<divup(64 * 1024, 256), 256, 0, stream>>>(lw3, wl3p, 10, 64, 1024, 0);

  conv1_kernel<<<divup(B * 32 * 32 * 128, 256), 256, 0, stream>>>(
      x, cw[1], cb[1], bn(1, 0), bn(1, 1), bn(1, 2), bn(1, 3), Abuf, B, 32, 32, 128);

  auto conv = [&](const signed char* inp, const signed char* wb, const float* bias,
                  const float* g, const float* b2, const float* m2, const float* v2,
                  short* of, signed char* os, int H, int W, int C, int O, int mode) {
    int tiles = ((B * H * W) >> 5) * (O >> 6);
    int blocks = divup(tiles, 4);
    switch (C) {
      case 128: binconv_wmma<128><<<blocks, 128, 0, stream>>>(inp, wb, bias, g, b2, m2, v2, of, os, B, H, W, O, mode); break;
      case 256: binconv_wmma<256><<<blocks, 128, 0, stream>>>(inp, wb, bias, g, b2, m2, v2, of, os, B, H, W, O, mode); break;
      case 512: binconv_wmma<512><<<blocks, 128, 0, stream>>>(inp, wb, bias, g, b2, m2, v2, of, os, B, H, W, O, mode); break;
    }
  };

  conv(Abuf, wb2, cb[2], cb[2], cb[2], cb[2], cb[2], Fbuf, nullptr, 32, 32, 128, 128, 0);
  pool_bn_sign<<<divup(B * 16 * 16 * 128, 256), 256, 0, stream>>>(
      Fbuf, Bbuf, cb[2], bn(2, 0), bn(2, 1), bn(2, 2), bn(2, 3), B, 32, 32, 128);
  conv(Bbuf, wb3, cb[3], bn(3, 0), bn(3, 1), bn(3, 2), bn(3, 3), nullptr, Abuf, 16, 16, 128, 256, 1);
  conv(Abuf, wb4, cb[4], cb[4], cb[4], cb[4], cb[4], Fbuf, nullptr, 16, 16, 256, 256, 0);
  pool_bn_sign<<<divup(B * 8 * 8 * 256, 256), 256, 0, stream>>>(
      Fbuf, Bbuf, cb[4], bn(4, 0), bn(4, 1), bn(4, 2), bn(4, 3), B, 16, 16, 256);
  conv(Bbuf, wb5, cb[5], bn(5, 0), bn(5, 1), bn(5, 2), bn(5, 3), nullptr, Abuf, 8, 8, 256, 512, 1);
  conv(Abuf, wb6, cb[6], cb[6], cb[6], cb[6], cb[6], Fbuf, nullptr, 8, 8, 512, 512, 0);
  pool_bn_sign<<<divup(B * 4 * 4 * 512, 256), 256, 0, stream>>>(
      Fbuf, Bbuf, cb[6], bn(6, 0), bn(6, 1), bn(6, 2), bn(6, 3), B, 8, 8, 512);

  {
    int blocks1 = divup((B >> 5) * (1024 >> 6), 4);
    bingemm_wmma<<<blocks1, 128, 0, stream>>>(Bbuf, wl1p, lb1, bn(7, 0), bn(7, 1), bn(7, 2), bn(7, 3),
                                              nullptr, Abuf, B, 8192, 1024, 1024, 1);
    bingemm_wmma<<<blocks1, 128, 0, stream>>>(Abuf, wl2p, lb2, bn(8, 0), bn(8, 1), bn(8, 2), bn(8, 3),
                                              nullptr, Bbuf, B, 1024, 1024, 1024, 1);
    int blocks3 = divup((B >> 5) * 1, 4);
    bingemm_wmma<<<blocks3, 128, 0, stream>>>(Bbuf, wl3p, lb3, bn(9, 0), bn(9, 1), bn(9, 2), bn(9, 3),
                                              logits, nullptr, B, 1024, 64, 10, 0);
  }

  logsoftmax_kernel<<<divup(B, 128), 128, 0, stream>>>(logits, (float*)d_out, B, 10, 64);
}
